// MinkUNetBase_8418135900539
// MI455X (gfx1250) — compile-verified, matrix operand order checked by reading, not run on hardware
//
#include <hip/hip_runtime.h>
#include <hip/hip_bf16.h>
#include <stdint.h>

// ---------------------------------------------------------------------------
// MinkUNetBase on a dense 32^3 grid, batch=2, gfx1250 (MI455X).
// v3: weights pre-packed (one-time kernels) into the per-lane WMMA B-fragment
// bf16 layout, K/N zero-padded; activations kept as channel-padded bf16 by the
// producer kernels. The GEMM hot loop is pure b128 loads + 4x v_wmma per
// K-step (16x64 tile per wave), with no LDS staging, guards or conversions.
// ---------------------------------------------------------------------------

typedef __attribute__((ext_vector_type(16))) __bf16 v16bf;
typedef __attribute__((ext_vector_type(8)))  __bf16 v8bf;
typedef __attribute__((ext_vector_type(8)))  float  v8f;

#define BN_EPS 1e-5f

// round-to-nearest-even f32 -> bf16 (integer ops, deterministic)
__device__ __forceinline__ __bf16 f2bf(float f) {
  union { float f; uint32_t u; } v; v.f = f;
  uint32_t r = (v.u + 0x7FFFu + ((v.u >> 16) & 1u)) >> 16;
  uint16_t h = (uint16_t)r;
  __bf16 o; __builtin_memcpy(&o, &h, 2);
  return o;
}

// 16 contiguous bf16 (two 16B b128 loads) -> v16bf fragment
__device__ __forceinline__ v16bf ld_frag16(const __bf16* p) {
  v8bf lo = *(const v8bf*)p;
  v8bf hi = *(const v8bf*)(p + 8);
  return __builtin_shufflevector(lo, hi, 0, 1, 2, 3, 4, 5, 6, 7,
                                 8, 9, 10, 11, 12, 13, 14, 15);
}

// two 16B loads at a K-split (khalf already applied to p; runs at +0 and +16)
__device__ __forceinline__ v16bf ld_fragA(const __bf16* p, bool ok) {
  v8bf lo = {}, hi = {};
  if (ok) {
    lo = *(const v8bf*)p;
    hi = *(const v8bf*)(p + 16);
  }
  return __builtin_shufflevector(lo, hi, 0, 1, 2, 3, 4, 5, 6, 7,
                                 8, 9, 10, 11, 12, 13, 14, 15);
}

// ---------------------------------------------------------------------------
// Weight packer: f32 [taps, Cin, Cout] -> bf16 tiles in per-lane WMMA
// B-fragment order. Tile(tap, kb, nb) = 512 bf16 (32 lanes x 16 elems):
//   elem(lane, j) = W[tap][kb*32 + (j>=8)*16 + (lane>=16)*8 + (j&7)]
//                    [nb*16 + (lane&15)]          (zero outside Cin/Cout)
// Layout: ((tap*nkb + kb)*nbcnt + nb)*512 + lane*16 + j. One-time per layer.
// ---------------------------------------------------------------------------
__global__ void pack_wgt(const float* __restrict__ w, __bf16* __restrict__ out,
                         int Cin, int Cout, int nkb, int nbcnt, int total)
{
  const int i = blockIdx.x * 256 + threadIdx.x;
  if (i >= total) return;
  const int e = i & 511;
  int tile = i >> 9;
  const int nb = tile % nbcnt; tile /= nbcnt;
  const int kb = tile % nkb;
  const int tap = tile / nkb;
  const int lane = e >> 4, j = e & 15;
  const int ci = kb * 32 + ((j >> 3) << 4) + ((lane >> 4) << 3) + (j & 7);
  const int gn = nb * 16 + (lane & 15);
  const float v = (ci < Cin && gn < Cout)
                      ? w[((size_t)tap * Cin + ci) * Cout + gn] : 0.f;
  out[i] = f2bf(v);
}

// f32 NDHWC -> channel-padded bf16 (zero fill pad channels)
__global__ void pack_act(__bf16* __restrict__ y, const float* __restrict__ x,
                         int M, int C, int Cpad)
{
  const int i = blockIdx.x * 256 + threadIdx.x;
  if (i >= M * Cpad) return;
  const int m = i / Cpad, c = i % Cpad;
  y[i] = f2bf(c < C ? x[(size_t)m * C + c] : 0.f);
}

// ---------------------------------------------------------------------------
// Implicit-GEMM 3D conv. Wave tile: M=16 x N=64 (4 accumulators). Block =
// 8 waves = 128 rows. Activations: bf16, channel stride Cpad (mult of 32).
// Weights: pre-packed bf16 fragments. Hot loop: 2 predicated A b128 loads,
// 8 B b128 loads, 4 WMMAs, 1 prefetch. No LDS, no barriers, no guards.
// ---------------------------------------------------------------------------
__global__ __launch_bounds__(256) void conv3d_wmma(
    const __bf16* __restrict__ act, const __bf16* __restrict__ wpk,
    const float* __restrict__ bias, float* __restrict__ out,
    int NB, int ID, int IH, int IW, int Cpad,
    int OD, int OH, int OW, int Cout, int nbcnt,
    int ksz, int stride, int pad)
{
  const int lane = threadIdx.x & 31, wave = threadIdx.x >> 5;
  const int M  = NB * OD * OH * OW;
  const int m0 = (blockIdx.x * 8 + wave) * 16;
  if (m0 >= M) return;               // wave-uniform; M is a multiple of 16
  const int nb0   = blockIdx.y * 4;  // four 16-col tiles -> 64 N per wave
  const int nkb   = Cpad >> 5;
  const int khalf = (lane >> 4) << 3;

  int t = m0 + (lane & 15);          // this lane's A row (voxel)
  const int ow = t % OW; t /= OW;
  const int oh = t % OH; t /= OH;
  const int od = t % OD; t /= OD;
  const int nb = t;

  v8f acc[4] = {{}, {}, {}, {}};
  for (int kd = 0; kd < ksz; ++kd) {
    const int id = od * stride - pad + kd;
    for (int kh = 0; kh < ksz; ++kh) {
      const int ih = oh * stride - pad + kh;
      for (int kw = 0; kw < ksz; ++kw) {
        const int iw = ow * stride - pad + kw;
        const bool ok = ((unsigned)id < (unsigned)ID) &&
                        ((unsigned)ih < (unsigned)IH) &&
                        ((unsigned)iw < (unsigned)IW);
        const __bf16* ap =
            act + (size_t)(((nb * ID + id) * IH + ih) * IW + iw) * Cpad + khalf;
        const int tap = (kd * ksz + kh) * ksz + kw;
        const __bf16* bp =
            wpk + ((size_t)(tap * nkb) * nbcnt + nb0) * 512 + lane * 16;
        for (int kb = 0; kb < nkb; ++kb) {
          const v16bf a = ld_fragA(ap, ok);
#pragma unroll
          for (int tt = 0; tt < 4; ++tt) {
            const v16bf b = ld_frag16(bp + tt * 512);
            acc[tt] = __builtin_amdgcn_wmma_f32_16x16x32_bf16(
                false, a, false, b, (short)0, acc[tt], false, false);
          }
          ap += 32;
          bp += (size_t)nbcnt * 512;
          __builtin_prefetch(bp, 0, 1);   // next K-step's B tiles
        }
      }
    }
  }

#pragma unroll
  for (int r = 0; r < 8; ++r) {
    const int m = m0 + ((lane >> 4) << 3) + r;   // C/D layout row
#pragma unroll
    for (int tt = 0; tt < 4; ++tt) {
      const int n = (nb0 + tt) * 16 + (lane & 15);
      if (n < Cout)
        out[(size_t)m * Cout + n] = acc[tt][r] + (bias ? bias[n] : 0.f);
    }
  }
}

// ---------------------------------------------------------------------------
// k=2,s=2 transposed conv = 8 independent 1x1 GEMMs over the coarse grid
// (blockIdx.z = parity octant). A rows contiguous & always valid.
// ---------------------------------------------------------------------------
__global__ __launch_bounds__(256) void convtr3d_wmma(
    const __bf16* __restrict__ act, const __bf16* __restrict__ wpk,
    float* __restrict__ out,
    int NB, int ID, int IH, int IW, int Cpad, int Cout, int nbcnt)
{
  const int lane = threadIdx.x & 31, wave = threadIdx.x >> 5;
  const int M  = NB * ID * IH * IW;
  const int m0 = (blockIdx.x * 8 + wave) * 16;
  if (m0 >= M) return;
  const int oct = blockIdx.z;
  const int pd = (oct >> 2) & 1, ph = (oct >> 1) & 1, pw = oct & 1;
  const int nb0   = blockIdx.y * 4;
  const int nkb   = Cpad >> 5;
  const int khalf = (lane >> 4) << 3;

  const __bf16* ap = act + (size_t)(m0 + (lane & 15)) * Cpad + khalf;
  const __bf16* bp =
      wpk + ((size_t)(oct * nkb) * nbcnt + nb0) * 512 + lane * 16;

  v8f acc[4] = {{}, {}, {}, {}};
  for (int kb = 0; kb < nkb; ++kb) {
    const v16bf a = ld_fragA(ap, true);
#pragma unroll
    for (int tt = 0; tt < 4; ++tt) {
      const v16bf b = ld_frag16(bp + tt * 512);
      acc[tt] = __builtin_amdgcn_wmma_f32_16x16x32_bf16(
          false, a, false, b, (short)0, acc[tt], false, false);
    }
    ap += 32;
    bp += (size_t)nbcnt * 512;
    __builtin_prefetch(bp, 0, 1);
  }

  const int OD = ID * 2, OH = IH * 2, OW = IW * 2;
#pragma unroll
  for (int r = 0; r < 8; ++r) {
    const int m = m0 + ((lane >> 4) << 3) + r;
    int tt2 = m;
    const int iw = tt2 % IW; tt2 /= IW;
    const int ih = tt2 % IH; tt2 /= IH;
    const int id = tt2 % ID; tt2 /= ID;
    const int nbb = tt2;
    const size_t obase =
        (size_t)(((nbb * OD + id * 2 + pd) * OH + ih * 2 + ph) * OW +
                 iw * 2 + pw) * Cout;
#pragma unroll
    for (int tt = 0; tt < 4; ++tt) {
      const int n = (nb0 + tt) * 16 + (lane & 15);
      if (n < Cout) out[obase + n] = acc[tt][r];
    }
  }
}

// ---------------------------------------------------------------------------
// Per-channel batch stats (train-mode BN): LDS tree reduction.
// ---------------------------------------------------------------------------
__global__ __launch_bounds__(256) void bn_stats(
    const float* __restrict__ x, int M, int C,
    float* __restrict__ mean, float* __restrict__ var)
{
  __shared__ float s1[256];
  __shared__ float s2[256];
  const int c = blockIdx.x;
  float sum = 0.f, sq = 0.f;
  for (int i = threadIdx.x; i < M; i += 256) {
    const float v = x[(size_t)i * C + c];
    sum += v; sq += v * v;
  }
  s1[threadIdx.x] = sum; s2[threadIdx.x] = sq;
  __syncthreads();
  for (int s = 128; s > 0; s >>= 1) {
    if (threadIdx.x < s) {
      s1[threadIdx.x] += s1[threadIdx.x + s];
      s2[threadIdx.x] += s2[threadIdx.x + s];
    }
    __syncthreads();
  }
  if (threadIdx.x == 0) {
    const float inv = 1.f / (float)M;
    const float mu = s1[0] * inv;
    mean[c] = mu;
    var[c]  = s2[0] * inv - mu * mu;
  }
}

// y = relu?( BN(x) [+ resid] ); writes f32 (dense C) + padded bf16 (Cpad)
__global__ void bn_apply(
    float* __restrict__ yf, __bf16* __restrict__ ybf,
    const float* __restrict__ x,
    const float* __restrict__ mean, const float* __restrict__ var,
    const float* __restrict__ g, const float* __restrict__ b,
    const float* __restrict__ resid, int M, int C, int Cpad, int relu)
{
  const int i = blockIdx.x * 256 + threadIdx.x;
  if (i >= M * Cpad) return;
  const int m = i / Cpad, c = i % Cpad;
  float v = 0.f;
  if (c < C) {
    v = (x[(size_t)m * C + c] - mean[c]) * rsqrtf(var[c] + BN_EPS) * g[c] + b[c];
    if (resid) v += resid[(size_t)m * C + c];
    if (relu) v = fmaxf(v, 0.f);
    yf[(size_t)m * C + c] = v;
  }
  ybf[i] = f2bf(v);
}

// channel concat of two f32 tensors -> bf16 (Ca+Cb is a multiple of 32 here)
__global__ void concat_bf(
    __bf16* __restrict__ y, const float* __restrict__ a,
    const float* __restrict__ b, int M, int Ca, int Cb)
{
  const int C = Ca + Cb;
  const int i = blockIdx.x * 256 + threadIdx.x;
  if (i >= M * C) return;
  const int m = i / C, c = i % C;
  const float v = (c < Ca) ? a[(size_t)m * Ca + c] : b[(size_t)m * Cb + (c - Ca)];
  y[i] = f2bf(v);
}

// ---------------------------------------------------------------------------
// Host orchestration
// ---------------------------------------------------------------------------
namespace {
struct CB  { const float* b; const float* g; const float* w; };
struct PW  { __bf16* p; int nkb; int nbcnt; };
struct Ten { float* f; __bf16* bf; int D; int C; int Cpad; };
}

extern "C" void kernel_launch(void* const* d_in, const int* in_sizes, int n_in,
                              void* d_out, int out_size, void* d_ws, size_t ws_size,
                              hipStream_t stream) {
  (void)in_sizes; (void)n_in; (void)out_size;
  const int NB = 2;

  // --- parse params in JAX pytree (sorted-dict) order -----------------------
  int ip = 0;
  auto nextCB = [&]() {
    CB c;
    c.b = (const float*)d_in[ip++];
    c.g = (const float*)d_in[ip++];
    c.w = (const float*)d_in[ip++];
    return c;
  };
  CB blk[8][5] = {};
  bool has_ds[8];
  for (int i = 0; i < 8; ++i) {
    has_ds[i] = (i != 0);                 // block1 is 32->32 (no downsample)
    blk[i][0] = nextCB();                 // bb0.conv1
    blk[i][1] = nextCB();                 // bb0.conv2
    if (has_ds[i]) blk[i][2] = nextCB();  // bb0.ds
    blk[i][3] = nextCB();                 // bb1.conv1
    blk[i][4] = nextCB();                 // bb1.conv2
  }
  CB conv0 = nextCB(), conv1 = nextCB(), conv2 = nextCB(),
     conv3 = nextCB(), conv4 = nextCB();
  CB ctr4 = nextCB(), ctr5 = nextCB(), ctr6 = nextCB(), ctr7 = nextCB();
  const float* final_bias = (const float*)d_in[ip++];
  const float* final_w    = (const float*)d_in[ip++];
  const float* x_in       = (const float*)d_in[ip++];

  // --- allocator: persistent region first, transient arena wraps above it ---
  char* wsbase = (char*)d_ws;
  size_t wsoff = 0, wsfloor = 0;
  auto balloc = [&](size_t bytes) -> void* {
    bytes = (bytes + 255) & ~(size_t)255;
    if (wsoff + bytes > ws_size) wsoff = wsfloor;   // bounded transient arena
    void* r = wsbase + wsoff;
    wsoff += bytes;
    return r;
  };
  float* mean_buf = (float*)balloc(512 * 4);
  float* var_buf  = (float*)balloc(512 * 4);

  // pack a weight tensor into WMMA B-fragment tiles (one-time, persistent)
  auto packW = [&](CB cb, int taps, int Cin, int Cout) -> PW {
    PW r;
    r.nkb   = (Cin + 31) / 32;
    r.nbcnt = 4 * ((Cout + 63) / 64);   // padded so every wave tile is backed
    const int total = taps * r.nkb * r.nbcnt * 512;
    r.p = (__bf16*)balloc((size_t)total * 2);
    pack_wgt<<<(total + 255) / 256, 256, 0, stream>>>(
        cb.w, r.p, Cin, Cout, r.nkb, r.nbcnt, total);
    return r;
  };

  // pack all weights (persistent region)
  const int cins[8]  = {32, 32, 64, 128, 384, 192, 128, 128};
  const int couts[8] = {32, 64, 128, 256, 256, 128, 96, 96};
  PW pblk[8][5];
  for (int i = 0; i < 8; ++i) {
    pblk[i][0] = packW(blk[i][0], 27, cins[i],  couts[i]);
    pblk[i][1] = packW(blk[i][1], 27, couts[i], couts[i]);
    if (has_ds[i]) pblk[i][2] = packW(blk[i][2], 1, cins[i], couts[i]);
    pblk[i][3] = packW(blk[i][3], 27, couts[i], couts[i]);
    pblk[i][4] = packW(blk[i][4], 27, couts[i], couts[i]);
  }
  PW pconv0 = packW(conv0, 1, 4, 32);
  PW pconv1 = packW(conv1, 8, 32, 32);
  PW pconv2 = packW(conv2, 8, 32, 32);
  PW pconv3 = packW(conv3, 8, 64, 64);
  PW pconv4 = packW(conv4, 8, 128, 128);
  PW pctr4  = packW(ctr4, 8, 256, 256);
  PW pctr5  = packW(ctr5, 8, 256, 128);
  PW pctr6  = packW(ctr6, 8, 128, 96);
  PW pctr7  = packW(ctr7, 8, 96, 96);
  CB finalCB{nullptr, nullptr, final_w};
  PW pfinal = packW(finalCB, 1, 96, 20);

  auto makeTen = [&](int D, int C) -> Ten {
    Ten t;
    t.D = D; t.C = C; t.Cpad = (C + 31) & ~31;
    const size_t M = (size_t)NB * D * D * D;
    t.f  = (float*)balloc(M * C * 4);
    t.bf = (__bf16*)balloc(M * t.Cpad * 2);
    return t;
  };
  // persistent skip tensors
  Ten p1 = makeTen(32, 32), b1 = makeTen(16, 32),
      b2 = makeTen(8, 64),  b3 = makeTen(4, 128);
  wsfloor = wsoff;   // everything above is transient

  // --- launch helpers -------------------------------------------------------
  auto launch_conv = [&](const __bf16* act, int D, int Cpad, PW pw,
                         const float* bias, float* out, int OD, int Cout,
                         int k, int s, int p) {
    const int M = NB * OD * OD * OD;
    dim3 grid((M + 127) / 128, (Cout + 63) / 64, 1);
    conv3d_wmma<<<grid, 256, 0, stream>>>(act, pw.p, bias, out,
                                          NB, D, D, D, Cpad,
                                          OD, OD, OD, Cout, pw.nbcnt, k, s, p);
  };
  auto bn = [&](Ten y, const float* conv_out, CB p, const float* resid,
                int M, int relu) {
    bn_stats<<<y.C, 256, 0, stream>>>(conv_out, M, y.C, mean_buf, var_buf);
    const int tot = M * y.Cpad;
    bn_apply<<<(tot + 255) / 256, 256, 0, stream>>>(
        y.f, y.bf, conv_out, mean_buf, var_buf, p.g, p.b, resid,
        M, y.C, y.Cpad, relu);
  };
  auto cblock = [&](Ten x, CB p, PW pw, int Cout, int k, int s, int pad,
                    int relu, const float* resid, Ten* dst) -> Ten {
    const int OD = (s == 2) ? x.D / 2 : x.D;
    const int M = NB * OD * OD * OD;
    float* co = (float*)balloc((size_t)M * Cout * 4);
    launch_conv(x.bf, x.D, x.Cpad, pw, nullptr, co, OD, Cout, k, s, pad);
    Ten y = dst ? *dst : makeTen(OD, Cout);
    bn(y, co, p, resid, M, relu);
    return y;
  };
  auto bblock = [&](Ten x, CB c1, CB c2, PW w1, PW w2,
                    const CB* ds, const PW* dsw, int Cout, Ten* dst) -> Ten {
    Ten t1 = cblock(x, c1, w1, Cout, 3, 1, 1, 1, nullptr, nullptr);
    const float* resid = x.f;
    if (ds) {
      Ten idn = cblock(x, *ds, *dsw, Cout, 1, 1, 0, 0, nullptr, nullptr);
      resid = idn.f;
    }
    return cblock(t1, c2, w2, Cout, 3, 1, 1, 1, resid, dst);
  };
  auto rlayer = [&](Ten x, CB* cb, PW* pw, bool ds, int Cout, Ten* dst) -> Ten {
    Ten y = bblock(x, cb[0], cb[1], pw[0], pw[1],
                   ds ? &cb[2] : nullptr, ds ? &pw[2] : nullptr, Cout, nullptr);
    return bblock(y, cb[3], cb[4], pw[3], pw[4], nullptr, nullptr, Cout, dst);
  };
  auto convtr = [&](Ten x, CB p, PW pw, int Cout) -> Ten {
    const int OD = x.D * 2;
    const int Min = NB * x.D * x.D * x.D;
    const int Mout = NB * OD * OD * OD;
    float* co = (float*)balloc((size_t)Mout * Cout * 4);
    dim3 grid((Min + 127) / 128, (Cout + 63) / 64, 8);
    convtr3d_wmma<<<grid, 256, 0, stream>>>(x.bf, pw.p, co,
                                            NB, x.D, x.D, x.D, x.Cpad,
                                            Cout, pw.nbcnt);
    Ten y = makeTen(OD, Cout);
    bn(y, co, p, nullptr, Mout, 1);
    return y;
  };
  auto concat = [&](Ten a, Ten b) -> Ten {
    const int M = NB * a.D * a.D * a.D;
    const int C = a.C + b.C;          // always a multiple of 32 here
    Ten y; y.D = a.D; y.C = C; y.Cpad = C; y.f = nullptr;
    y.bf = (__bf16*)balloc((size_t)M * C * 2);
    const int tot = M * C;
    concat_bf<<<(tot + 255) / 256, 256, 0, stream>>>(y.bf, a.f, b.f, M, a.C, b.C);
    return y;
  };

  // --- UNet forward ---------------------------------------------------------
  // pack raw input (C=4 -> Cpad=32 bf16)
  Ten x0; x0.D = 32; x0.C = 4; x0.Cpad = 32; x0.f = const_cast<float*>(x_in);
  {
    const int M = NB * 32 * 32 * 32, tot = M * 32;
    x0.bf = (__bf16*)balloc((size_t)tot * 2);
    pack_act<<<(tot + 255) / 256, 256, 0, stream>>>(x0.bf, x_in, M, 4, 32);
  }
  cblock(x0, conv0, pconv0, 32, 1, 1, 0, 1, nullptr, &p1);       // full, 32
  Ten o = cblock(p1, conv1, pconv1, 32, 2, 2, 0, 1, nullptr, nullptr);  // /2
  rlayer(o, blk[0], pblk[0], false, 32, &b1);                    // /2, 32
  o = cblock(b1, conv2, pconv2, 32, 2, 2, 0, 1, nullptr, nullptr);      // /4
  rlayer(o, blk[1], pblk[1], true, 64, &b2);                     // /4, 64
  o = cblock(b2, conv3, pconv3, 64, 2, 2, 0, 1, nullptr, nullptr);      // /8
  rlayer(o, blk[2], pblk[2], true, 128, &b3);                    // /8, 128
  o = cblock(b3, conv4, pconv4, 128, 2, 2, 0, 1, nullptr, nullptr);     // /16
  o = rlayer(o, blk[3], pblk[3], true, 256, nullptr);            // /16, 256
  o = convtr(o, ctr4, pctr4, 256);                               // /8
  o = rlayer(concat(o, b3), blk[4], pblk[4], true, 256, nullptr);// /8, 256
  o = convtr(o, ctr5, pctr5, 128);                               // /4
  o = rlayer(concat(o, b2), blk[5], pblk[5], true, 128, nullptr);// /4, 128
  o = convtr(o, ctr6, pctr6, 96);                                // /2
  o = rlayer(concat(o, b1), blk[6], pblk[6], true, 96, nullptr); // /2, 96
  o = convtr(o, ctr7, pctr7, 96);                                // full
  o = rlayer(concat(o, p1), blk[7], pblk[7], true, 96, nullptr); // full, 96
  // final 1x1 conv + bias directly into d_out (Cout=20, padded pack + guards)
  launch_conv(o.bf, 32, o.Cpad, pfinal, final_bias, (float*)d_out,
              32, 20, 1, 1, 0);
}
